// Transformer_34016140984664
// MI455X (gfx1250) — matrix operand order repaired and checked by reading, not compiled, not run on hardware
//
#include <hip/hip_runtime.h>

// ---------------------------------------------------------------------------
// CDNA5 (gfx1250) GPT forward pass: bf16 WMMA GEMMs + flash attention.
// wave32, WGP mode. Matrix math via v_wmma_f32_16x16x32_bf16; GEMM global->LDS
// staging via async-tensor path (global_load_async_to_lds_b128 + ASYNCcnt),
// double-buffered so the copy of tile k+1 overlaps WMMA of tile k.
// ---------------------------------------------------------------------------

typedef __attribute__((ext_vector_type(16))) __bf16       v16bf;
typedef __attribute__((ext_vector_type(8)))  float        v8f;
typedef __attribute__((ext_vector_type(8)))  unsigned int v8u;
typedef __attribute__((ext_vector_type(4)))  unsigned int v4u;

#define WMMA_BF16(a, b, c) \
    __builtin_amdgcn_wmma_f32_16x16x32_bf16(false, (a), false, (b), (short)0, (c), false, false)

union Frag { v4u h4[2]; v16bf v; };

__device__ __forceinline__ __bf16 f2bf(float f) {
    unsigned int u = __builtin_bit_cast(unsigned int, f);
    u += 0x7FFFu + ((u >> 16) & 1u);                 // round-to-nearest-even
    unsigned short s = (unsigned short)(u >> 16);
    return __builtin_bit_cast(__bf16, s);
}

__device__ __forceinline__ v8f v8f_zero() {
    v8f z = {0.f, 0.f, 0.f, 0.f, 0.f, 0.f, 0.f, 0.f};
    return z;
}

// Async global->LDS 16-byte copy (CDNA5 TDM-lite path, tracked by ASYNCcnt).
// LDS byte address = low 32 bits of the generic pointer (LDS aperture rule).
__device__ __forceinline__ void async_copy16(const __bf16* gsrc, __bf16* lds_dst) {
    unsigned int       l = (unsigned int)(unsigned long long)(uintptr_t)lds_dst;
    unsigned long long g = (unsigned long long)(uintptr_t)gsrc;
    asm volatile("global_load_async_to_lds_b128 %0, %1, off"
                 :: "v"(l), "v"(g) : "memory");
}
__device__ __forceinline__ void wait_async() {
    asm volatile("s_wait_asynccnt 0x0" ::: "memory");
}

// ---------------------------------------------------------------------------
// f32 -> bf16 conversion (weights, once per launch)
// ---------------------------------------------------------------------------
__global__ __launch_bounds__(256) void k_f32_to_bf16(const float* __restrict__ in,
                                                     __bf16* __restrict__ out, size_t n) {
    size_t i = (size_t)blockIdx.x * blockDim.x + threadIdx.x;
    size_t stride = (size_t)gridDim.x * blockDim.x;
    for (; i < n; i += stride) out[i] = f2bf(in[i]);
}

// ---------------------------------------------------------------------------
// x = tok_emb[idx] + pos_emb   (f32 residual stream)
// ---------------------------------------------------------------------------
__global__ __launch_bounds__(256) void k_embed(const int* __restrict__ idx,
                                               const float* __restrict__ tok,
                                               const float* __restrict__ pos,
                                               float* __restrict__ x,
                                               int T_, int E_, size_t n) {
    size_t i = (size_t)blockIdx.x * blockDim.x + threadIdx.x;
    size_t stride = (size_t)gridDim.x * blockDim.x;
    for (; i < n; i += stride) {
        size_t row = i / (size_t)E_;
        int e = (int)(i - row * (size_t)E_);
        int t = (int)(row % (size_t)T_);
        x[i] = tok[(size_t)idx[row] * E_ + e] + pos[(size_t)t * E_ + e];
    }
}

// ---------------------------------------------------------------------------
// LayerNorm: f32 [rows,E] -> bf16 [rows,E]   (one block per row)
// ---------------------------------------------------------------------------
__global__ __launch_bounds__(256) void k_layernorm(const float* __restrict__ x,
                                                   const float* __restrict__ g,
                                                   const float* __restrict__ bta,
                                                   __bf16* __restrict__ out, int E_) {
    const int row = blockIdx.x;
    const float* xr = x + (size_t)row * E_;
    __shared__ float rs[8], rs2[8];
    const int lane = threadIdx.x & 31, wv = threadIdx.x >> 5;
    float s = 0.f, s2 = 0.f;
    for (int i = threadIdx.x; i < E_; i += 256) { float v = xr[i]; s += v; s2 += v * v; }
    for (int o = 16; o > 0; o >>= 1) { s += __shfl_xor(s, o, 32); s2 += __shfl_xor(s2, o, 32); }
    if (lane == 0) { rs[wv] = s; rs2[wv] = s2; }
    __syncthreads();
    float ts = 0.f, ts2 = 0.f;
    for (int w = 0; w < 8; ++w) { ts += rs[w]; ts2 += rs2[w]; }
    const float mu  = ts / E_;
    const float var = ts2 / E_ - mu * mu;
    const float inv = rsqrtf(var + 1e-5f);
    for (int i = threadIdx.x; i < E_; i += 256)
        out[(size_t)row * E_ + i] = f2bf((xr[i] - mu) * inv * g[i] + bta[i]);
}

// ---------------------------------------------------------------------------
// bf16 WMMA GEMM:  C[M,N] = A[M,K] @ B[K,N] (+bias +residual, optional ReLU)
// Block tile 128x128, BK=32, 256 threads = 8 waves, each wave 2x4 WMMA tiles.
// Double-buffered LDS:
//   A tile  : async global->LDS (b128 chunks), no VGPR transit, ASYNCcnt.
//   B tile  : register-pipelined + transposed ds_store scatter to [n][k] so
//             B-fragments are contiguous ds_load_b128 pairs.
// Fragment layouts per ISA: A k = (e&7)+8*half+16*(e>>3), B k = same as A.
// ---------------------------------------------------------------------------
__global__ __launch_bounds__(256) void k_gemm_bf16(
    const __bf16* __restrict__ A, const __bf16* __restrict__ Bw,
    float* Cf, __bf16* Cb,
    const float* __restrict__ bias, const float* resid,
    int M, int N, int K, int relu) {
    __shared__ __align__(16) __bf16 As[2][128][40];
    __shared__ __align__(16) __bf16 Bs[2][128][40];

    const int tid  = threadIdx.x;
    const int lane = tid & 31, wave = tid >> 5;
    const int half = lane >> 4, l16 = lane & 15;
    const int m0 = (wave & 3) * 32;   // wave row offset inside block tile
    const int n0 = (wave >> 2) * 64;  // wave col offset inside block tile
    const int bm = blockIdx.y * 128, bn = blockIdx.x * 128;
    // A async-copy mapping: 512 x 16B chunks, 2 per thread (rows r and r+64)
    const int ar = tid >> 2, acp = (tid & 3) * 8;
    // B tile mapping: 32x128, 16 elems (32B) per thread
    const int brow = tid >> 3, bcol = (tid & 7) * 16;

    v8f acc[2][4];
#pragma unroll
    for (int i = 0; i < 2; ++i)
#pragma unroll
        for (int j = 0; j < 4; ++j) acc[i][j] = v8f_zero();

    // --- prologue: stage tile 0 into buffer 0 ---
    async_copy16(A + (size_t)(bm + ar) * K + acp,      &As[0][ar][acp]);
    async_copy16(A + (size_t)(bm + ar + 64) * K + acp, &As[0][ar + 64][acp]);
    {
        v8u d = *(const v8u*)(Bw + (size_t)brow * N + bn + bcol);
        const __bf16* e = (const __bf16*)&d;
#pragma unroll
        for (int j = 0; j < 16; ++j) Bs[0][bcol + j][brow] = e[j];
    }
    wait_async();
    __syncthreads();

    const int nk = K / 32;
    for (int ki = 0; ki < nk; ++ki) {
        const int cur = ki & 1, nxt = cur ^ 1;
        const bool more = (ki + 1 < nk);
        v8u breg;
        if (more) {
            const int k1 = (ki + 1) * 32;
            // async copy of next A tile overlaps the WMMAs below
            async_copy16(A + (size_t)(bm + ar) * K + k1 + acp,      &As[nxt][ar][acp]);
            async_copy16(A + (size_t)(bm + ar + 64) * K + k1 + acp, &As[nxt][ar + 64][acp]);
            // next B tile into registers (consumed after compute)
            breg = *(const v8u*)(Bw + (size_t)(k1 + brow) * N + bn + bcol);
            if (ki + 2 < nk)
                __builtin_prefetch(Bw + (size_t)((ki + 2) * 32 + brow) * N + bn + bcol, 0, 1);
        }

        // --- compute on buffer `cur` ---
        Frag af[2];
#pragma unroll
        for (int im = 0; im < 2; ++im) {
            const int mr = m0 + im * 16 + l16;
            af[im].h4[0] = *(const v4u*)&As[cur][mr][8 * half];        // k = 8h..8h+7
            af[im].h4[1] = *(const v4u*)&As[cur][mr][16 + 8 * half];   // k = 16+8h..23+8h
        }
#pragma unroll
        for (int jn = 0; jn < 4; ++jn) {
            const int nr = n0 + jn * 16 + l16;
            Frag bfr;
            bfr.h4[0] = *(const v4u*)&Bs[cur][nr][8 * half];           // k = 8h..8h+7
            bfr.h4[1] = *(const v4u*)&Bs[cur][nr][16 + 8 * half];      // k = 16+8h..23+8h
#pragma unroll
            for (int im = 0; im < 2; ++im)
                acc[im][jn] = WMMA_BF16(af[im].v, bfr.v, acc[im][jn]);
        }

        if (more) {
            const __bf16* e = (const __bf16*)&breg;
#pragma unroll
            for (int j = 0; j < 16; ++j) Bs[nxt][bcol + j][brow] = e[j];
            wait_async();   // next A tile fully landed in LDS
        }
        __syncthreads();    // everyone done reading cur / writing nxt
    }

    // Epilogue: D layout -> lane(l16)=col, vgpr r + 8*half = row.
#pragma unroll
    for (int jn = 0; jn < 4; ++jn) {
        const int gn = bn + n0 + jn * 16 + l16;
        const float bv = bias ? bias[gn] : 0.f;
#pragma unroll
        for (int im = 0; im < 2; ++im) {
#pragma unroll
            for (int r = 0; r < 8; ++r) {
                const int gm = bm + m0 + im * 16 + r + 8 * half;
                float v = acc[im][jn][r] + bv;
                if (resid) v += resid[(size_t)gm * N + gn];
                if (relu)  v = fmaxf(v, 0.f);
                if (Cf) Cf[(size_t)gm * N + gn] = v;
                if (Cb) Cb[(size_t)gm * N + gn] = f2bf(v);
            }
        }
    }
}

// ---------------------------------------------------------------------------
// Flash attention (causal). Grid (T/128, H, B), 256 threads = 8 waves.
// Each wave owns a 16-query tile; sweeps key blocks of 32 with online softmax.
// S = Q@K^T via two k=32 WMMAs; P goes C-layout -> A-layout via per-wave LDS
// bounce guarded by s_wait_dscnt. V staged LDS-transposed cooperatively.
// ---------------------------------------------------------------------------
__global__ __launch_bounds__(256) void k_attention(
    const __bf16* __restrict__ qb, const __bf16* __restrict__ kb,
    const __bf16* __restrict__ vb, __bf16* __restrict__ ob,
    int T_, int E_) {
    const int bz = blockIdx.z, h = blockIdx.y, qt = blockIdx.x;
    const int tid = threadIdx.x, lane = tid & 31, wave = tid >> 5;
    const int half = lane >> 4, l16 = lane & 15;
    const int q0 = qt * 128 + wave * 16;
    const size_t rowbase = (size_t)bz * T_;
    const int hc = h * 64;

    __shared__ __align__(16) __bf16 Vs[64][40];       // V tile, [dim][key]
    __shared__ __align__(16) __bf16 Ps[8][16][40];    // per-wave P transpose bounce

    // Q A-fragments (dims 0..31 / 32..63), straight from global (contiguous)
    Frag qa[2];
    {
        const __bf16* qr = qb + (rowbase + q0 + l16) * E_ + hc;
#pragma unroll
        for (int f = 0; f < 2; ++f) {
            qa[f].h4[0] = *(const v4u*)(qr + f * 32 + 8 * half);
            qa[f].h4[1] = *(const v4u*)(qr + f * 32 + 16 + 8 * half);
        }
    }

    v8f oacc[4];
#pragma unroll
    for (int j = 0; j < 4; ++j) oacc[j] = v8f_zero();
    float mrow[8], lrow[8];
#pragma unroll
    for (int r = 0; r < 8; ++r) { mrow[r] = -1e30f; lrow[r] = 0.f; }

    const int kend = qt * 128 + 128;
    for (int kb0 = 0; kb0 < kend; kb0 += 32) {
        __syncthreads();
        {   // stage V[kb0..kb0+31][hc..hc+63] transposed into Vs[dim][key]
            const int kk = tid >> 3;
            const int dc = (tid & 7) * 8;
            v4u d = *(const v4u*)(vb + (rowbase + kb0 + kk) * E_ + hc + dc);
            const __bf16* e8 = (const __bf16*)&d;
#pragma unroll
            for (int j = 0; j < 8; ++j) Vs[dc + j][kk] = e8[j];
        }
        __syncthreads();
        if (kb0 < q0 + 16) {   // causal: this key block intersects wave's range
            v8f st[2];
#pragma unroll
            for (int kt = 0; kt < 2; ++kt) {
                v8f s = v8f_zero();
                // K^T B-fragments read directly from global (row of K is contiguous)
                const __bf16* kr = kb + (rowbase + kb0 + kt * 16 + l16) * E_ + hc;
#pragma unroll
                for (int f = 0; f < 2; ++f) {
                    Frag bfr;
                    bfr.h4[0] = *(const v4u*)(kr + f * 32 + 8 * half);
                    bfr.h4[1] = *(const v4u*)(kr + f * 32 + 16 + 8 * half);
                    s = WMMA_BF16(qa[f].v, bfr.v, s);
                }
                st[kt] = s;
            }
            float p0[8], p1[8], alpha[8];
#pragma unroll
            for (int r = 0; r < 8; ++r) {
                const int qm = q0 + r + 8 * half;
                const int kn0 = kb0 + l16, kn1 = kb0 + 16 + l16;
                float s0 = st[0][r] * 0.125f;   // HS^-0.5 = 1/8
                float s1 = st[1][r] * 0.125f;
                if (kn0 > qm) s0 = -1e30f;
                if (kn1 > qm) s1 = -1e30f;
                float mx = fmaxf(s0, s1);
                for (int o = 1; o < 16; o <<= 1) mx = fmaxf(mx, __shfl_xor(mx, o, 16));
                const float mn = fmaxf(mrow[r], mx);
                const float a  = __expf(mrow[r] - mn);
                const float e0 = __expf(s0 - mn);
                const float e1 = __expf(s1 - mn);
                float rsum = e0 + e1;
                for (int o = 1; o < 16; o <<= 1) rsum += __shfl_xor(rsum, o, 16);
                lrow[r] = lrow[r] * a + rsum;
                mrow[r] = mn;
                alpha[r] = a;
                p0[r] = e0; p1[r] = e1;
            }
#pragma unroll
            for (int j = 0; j < 4; ++j)
#pragma unroll
                for (int r = 0; r < 8; ++r) oacc[j][r] *= alpha[r];

            // P: C-layout -> LDS -> A-layout (per-wave private region)
#pragma unroll
            for (int r = 0; r < 8; ++r) {
                Ps[wave][r + 8 * half][l16]      = f2bf(p0[r]);
                Ps[wave][r + 8 * half][16 + l16] = f2bf(p1[r]);
            }
            asm volatile("s_wait_dscnt 0x0" ::: "memory");
            Frag pf;
            pf.h4[0] = *(const v4u*)&Ps[wave][l16][8 * half];
            pf.h4[1] = *(const v4u*)&Ps[wave][l16][16 + 8 * half];
#pragma unroll
            for (int j = 0; j < 4; ++j) {
                Frag vf;
                const int nd = j * 16 + l16;
                vf.h4[0] = *(const v4u*)&Vs[nd][8 * half];
                vf.h4[1] = *(const v4u*)&Vs[nd][16 + 8 * half];
                oacc[j] = WMMA_BF16(pf.v, vf.v, oacc[j]);
            }
        }
    }
    // O = acc / l, store bf16 (concat-head layout [row, E])
#pragma unroll
    for (int j = 0; j < 4; ++j)
#pragma unroll
        for (int r = 0; r < 8; ++r) {
            const size_t grow = rowbase + q0 + r + 8 * half;
            ob[grow * E_ + hc + j * 16 + l16] = f2bf(oacc[j][r] / lrow[r]);
        }
}

// ---------------------------------------------------------------------------
// Per-row NLL: nll[row] = logsumexp(logits[row,:]) - logits[row, target]
// ---------------------------------------------------------------------------
__global__ __launch_bounds__(256) void k_nll_rows(const float* logits,
                                                  const int* __restrict__ targets,
                                                  float* __restrict__ nll, int Vv) {
    const int row = blockIdx.x;
    const float* lr = logits + (size_t)row * Vv;
    __shared__ float red[8];
    const int lane = threadIdx.x & 31, wv = threadIdx.x >> 5;
    float mx = -1e30f;
    for (int i = threadIdx.x; i < Vv; i += 256) mx = fmaxf(mx, lr[i]);
    for (int o = 16; o > 0; o >>= 1) mx = fmaxf(mx, __shfl_xor(mx, o, 32));
    if (lane == 0) red[wv] = mx;
    __syncthreads();
    float m = red[0];
    for (int w = 1; w < 8; ++w) m = fmaxf(m, red[w]);
    __syncthreads();
    float s = 0.f;
    for (int i = threadIdx.x; i < Vv; i += 256) s += __expf(lr[i] - m);
    for (int o = 16; o > 0; o >>= 1) s += __shfl_xor(s, o, 32);
    if (lane == 0) red[wv] = s;
    __syncthreads();
    if (threadIdx.x == 0) {
        float tot = 0.f;
        for (int w = 0; w < 8; ++w) tot += red[w];
        nll[row] = logf(tot) + m - lr[targets[row]];
    }
}

__global__ __launch_bounds__(256) void k_nll_reduce(const float* __restrict__ nll,
                                                    float* __restrict__ out, int n) {
    __shared__ float red[8];
    const int lane = threadIdx.x & 31, wv = threadIdx.x >> 5;
    float s = 0.f;
    for (int i = threadIdx.x; i < n; i += 256) s += nll[i];
    for (int o = 16; o > 0; o >>= 1) s += __shfl_xor(s, o, 32);
    if (lane == 0) red[wv] = s;
    __syncthreads();
    if (threadIdx.x == 0) {
        float tot = 0.f;
        for (int w = 0; w < 8; ++w) tot += red[w];
        out[0] = tot / (float)n;
    }
}

// ---------------------------------------------------------------------------
// Host launcher
// ---------------------------------------------------------------------------
extern "C" void kernel_launch(void* const* d_in, const int* in_sizes, int n_in,
                              void* d_out, int out_size, void* d_ws, size_t ws_size,
                              hipStream_t stream) {
    (void)in_sizes; (void)n_in; (void)out_size; (void)ws_size;
    const int*   idx     = (const int*)d_in[0];
    const int*   targets = (const int*)d_in[1];
    const float* tok_emb = (const float*)d_in[2];
    const float* pos_emb = (const float*)d_in[3];
    const float* Wq      = (const float*)d_in[4];
    const float* Wk      = (const float*)d_in[5];
    const float* Wv      = (const float*)d_in[6];
    const float* Wproj   = (const float*)d_in[7];
    const float* bproj   = (const float*)d_in[8];
    const float* ln1_g   = (const float*)d_in[9];
    const float* ln1_b   = (const float*)d_in[10];
    const float* ln2_g   = (const float*)d_in[11];
    const float* ln2_b   = (const float*)d_in[12];
    const float* W1      = (const float*)d_in[13];
    const float* b1      = (const float*)d_in[14];
    const float* W2      = (const float*)d_in[15];
    const float* b2      = (const float*)d_in[16];
    const float* lnf_g   = (const float*)d_in[17];
    const float* lnf_b   = (const float*)d_in[18];
    const float* Wlm     = (const float*)d_in[19];
    const float* blm     = (const float*)d_in[20];

    constexpr int Bb = 4, Tt = 1024, Ee = 1024, Hh = 16, Ll = 4, Vv = 32000;
    constexpr int Mr = Bb * Tt;   // 4096 rows
    constexpr int F  = 4 * Ee;    // 4096 MLP width

    float* logits = (float*)d_out;
    float* loss   = (float*)d_out + (size_t)Mr * Vv;

    char* ws = (char*)d_ws;
    size_t off = 0;
    auto carve = [&](size_t bytes) -> void* {
        void* p = (void*)(ws + off);
        off += (bytes + 255) & ~(size_t)255;
        return p;
    };
    __bf16* wq_bf  = (__bf16*)carve((size_t)Ll * Ee * Ee * 2);
    __bf16* wk_bf  = (__bf16*)carve((size_t)Ll * Ee * Ee * 2);
    __bf16* wv_bf  = (__bf16*)carve((size_t)Ll * Ee * Ee * 2);
    __bf16* wp_bf  = (__bf16*)carve((size_t)Ll * Ee * Ee * 2);
    __bf16* w1_bf  = (__bf16*)carve((size_t)Ll * Ee * F * 2);
    __bf16* w2_bf  = (__bf16*)carve((size_t)Ll * F * Ee * 2);
    __bf16* wlm_bf = (__bf16*)carve((size_t)Ee * Vv * 2);
    float*  x      = (float*) carve((size_t)Mr * Ee * 4);
    __bf16* hbf    = (__bf16*)carve((size_t)Mr * Ee * 2);
    __bf16* qbf    = (__bf16*)carve((size_t)Mr * Ee * 2);
    __bf16* kbf    = (__bf16*)carve((size_t)Mr * Ee * 2);
    __bf16* vbf    = (__bf16*)carve((size_t)Mr * Ee * 2);
    __bf16* abf    = (__bf16*)carve((size_t)Mr * Ee * 2);
    __bf16* mbf    = (__bf16*)carve((size_t)Mr * F * 2);
    float*  nllb   = (float*) carve((size_t)Mr * 4);

    dim3 blk(256);
    // Weight downconverts (deterministic, re-done every call)
    k_f32_to_bf16<<<2048, blk, 0, stream>>>(Wq,    wq_bf,  (size_t)Ll * Ee * Ee);
    k_f32_to_bf16<<<2048, blk, 0, stream>>>(Wk,    wk_bf,  (size_t)Ll * Ee * Ee);
    k_f32_to_bf16<<<2048, blk, 0, stream>>>(Wv,    wv_bf,  (size_t)Ll * Ee * Ee);
    k_f32_to_bf16<<<2048, blk, 0, stream>>>(Wproj, wp_bf,  (size_t)Ll * Ee * Ee);
    k_f32_to_bf16<<<2048, blk, 0, stream>>>(W1,    w1_bf,  (size_t)Ll * Ee * F);
    k_f32_to_bf16<<<2048, blk, 0, stream>>>(W2,    w2_bf,  (size_t)Ll * F * Ee);
    k_f32_to_bf16<<<2048, blk, 0, stream>>>(Wlm,   wlm_bf, (size_t)Ee * Vv);

    k_embed<<<2048, blk, 0, stream>>>(idx, tok_emb, pos_emb, x, Tt, Ee, (size_t)Mr * Ee);

    const dim3 gE(Ee / 128, Mr / 128);   // (8, 32)
    const dim3 gF(F  / 128, Mr / 128);   // (32, 32)
    const dim3 gA(Tt / 128, Hh, Bb);     // (8, 16, 4)

    for (int l = 0; l < Ll; ++l) {
        k_layernorm<<<Mr, blk, 0, stream>>>(x, ln1_g + (size_t)l * Ee, ln1_b + (size_t)l * Ee, hbf, Ee);
        k_gemm_bf16<<<gE, blk, 0, stream>>>(hbf, wq_bf + (size_t)l * Ee * Ee, nullptr, qbf,
                                            nullptr, nullptr, Mr, Ee, Ee, 0);
        k_gemm_bf16<<<gE, blk, 0, stream>>>(hbf, wk_bf + (size_t)l * Ee * Ee, nullptr, kbf,
                                            nullptr, nullptr, Mr, Ee, Ee, 0);
        k_gemm_bf16<<<gE, blk, 0, stream>>>(hbf, wv_bf + (size_t)l * Ee * Ee, nullptr, vbf,
                                            nullptr, nullptr, Mr, Ee, Ee, 0);
        k_attention<<<gA, blk, 0, stream>>>(qbf, kbf, vbf, abf, Tt, Ee);
        k_gemm_bf16<<<gE, blk, 0, stream>>>(abf, wp_bf + (size_t)l * Ee * Ee, x, nullptr,
                                            bproj + (size_t)l * Ee, x, Mr, Ee, Ee, 0);
        k_layernorm<<<Mr, blk, 0, stream>>>(x, ln2_g + (size_t)l * Ee, ln2_b + (size_t)l * Ee, hbf, Ee);
        k_gemm_bf16<<<gF, blk, 0, stream>>>(hbf, w1_bf + (size_t)l * Ee * F, nullptr, mbf,
                                            b1 + (size_t)l * F, nullptr, Mr, F, Ee, 1);
        k_gemm_bf16<<<gE, blk, 0, stream>>>(mbf, w2_bf + (size_t)l * F * Ee, x, nullptr,
                                            b2 + (size_t)l * Ee, x, Mr, Ee, F, 0);
    }

    k_layernorm<<<Mr, blk, 0, stream>>>(x, lnf_g, lnf_b, hbf, Ee);
    const dim3 gV(Vv / 128, Mr / 128);   // (250, 32)
    k_gemm_bf16<<<gV, blk, 0, stream>>>(hbf, wlm_bf, logits, nullptr, blm, nullptr,
                                        Mr, Vv, Ee, 0);
    k_nll_rows<<<Mr, blk, 0, stream>>>(logits, targets, nllb, Vv);
    k_nll_reduce<<<1, blk, 0, stream>>>(nllb, loss, Mr);
}
